// ResShift_41480794144829
// MI455X (gfx1250) — matrix operand order repaired and checked by reading, not compiled, not run on hardware
//
#include <hip/hip_runtime.h>
#include <hip/hip_bf16.h>
#include <stdint.h>

// ---------------------------------------------------------------------------
// MI455X (gfx1250) implementation of the DeepShift bottleneck block.
//
// Pixel-major f16 activations, fragment-packed f16 weights, WMMA
// f32_16x16x32_f16 GEMMs with global_load_async_to_lds_b128 double buffering,
// fused BN/ReLU/residual/absmax epilogues.
// This revision hoists all fragment loads ahead of the WMMA burst so the
// compiler emits one ds_load clause + one wait + 8 back-to-back v_wmma.
// ---------------------------------------------------------------------------

typedef _Float16 v16h __attribute__((ext_vector_type(16)));
typedef _Float16 v8h  __attribute__((ext_vector_type(8)));
typedef float    v8f  __attribute__((ext_vector_type(8)));

#define QMAX 32767.0f

__device__ __forceinline__ float qscale(float amax) {
    return fmaxf(amax, 1e-12f) * (1.0f / QMAX);
}

__device__ __forceinline__ float pow2q(float w) {
    // DeepShift: sign(w) * 2^clip(round(log2|w|), -14, 0)
    float a  = fabsf(w) + 1e-20f;
    float p  = rintf(log2f(a));
    p        = fminf(fmaxf(p, -14.0f), 0.0f);
    float sg = (w > 0.0f) ? 1.0f : ((w < 0.0f) ? -1.0f : 0.0f);
    return sg * exp2f(p);
}

// CDNA5 async copy: 16 bytes global -> LDS, per-lane addresses, ASYNCcnt.
__device__ __forceinline__ void async_cp16(unsigned lds_off, const void* gptr) {
    asm volatile("global_load_async_to_lds_b128 %0, %1, off"
                 :: "v"(lds_off), "v"(gptr)
                 : "memory");
}
__device__ __forceinline__ void wait_async0() {
    asm volatile("s_wait_asynccnt 0x0" ::: "memory");
}

__device__ __forceinline__ v16h cat16(v8h lo, v8h hi) {
    return __builtin_shufflevector(lo, hi, 0, 1, 2, 3, 4, 5, 6, 7,
                                   8, 9, 10, 11, 12, 13, 14, 15);
}

// ---------------------------------------------------------------------------
// Small prep kernels
// ---------------------------------------------------------------------------
__global__ void init_scales_k(float* s) {
    if (threadIdx.x < 8) s[threadIdx.x] = 0.0f;
}

__global__ void bn_prep_k(const float* g, const float* b, const float* m,
                          const float* v, float* out, int n) {
    int i = blockIdx.x * 256 + threadIdx.x;
    if (i < n) {
        float inv = g[i] * rsqrtf(v[i] + 1e-5f);
        out[2 * i]     = inv;
        out[2 * i + 1] = b[i] - m[i] * inv;
    }
}

__global__ void quant_pegw_k(const float* w, float* out, int n) {
    int i = blockIdx.x * 256 + threadIdx.x;
    if (i < n) out[i] = pow2q(w[i]);
}

// Pack pow2-quantized weights into WMMA B-fragment order.
// B element (k, n): kb=k/32, nb=n/16, lane=(n&15)|(((k>>4)&1)<<4), j=k&15
// linear = ((kb*(N/16)+nb)*32 + lane)*16 + j  -> per-lane v16h is contiguous.
__global__ __launch_bounds__(256)
void packw_k(const float* w, _Float16* out, int N, int K, int conv3x3) {
    int idx = blockIdx.x * 256 + threadIdx.x;
    if (idx >= N * K) return;
    int n = idx / K, k = idx - n * K;
    float v;
    if (conv3x3) {                       // w2: [oc][ic=256][3][3], im2col k = tap*256 + ic
        int c = k & 255, j = k >> 8;
        v = w[((size_t)n * 256 + c) * 9 + j];
    } else {
        v = w[(size_t)n * K + k];
    }
    v = pow2q(v);
    int kb = k >> 5, nb = n >> 4;
    int ln = (n & 15) | (((k >> 4) & 1) << 4);
    size_t o = ((size_t)(kb * (N >> 4) + nb) * 32 + ln) * 16 + (k & 15);
    out[o] = (_Float16)v;
}

__global__ __launch_bounds__(256)
void absmax_k(const float* x, long long n, float* slot) {
    __shared__ float red[256];
    float m = 0.0f;
    for (long long i = (long long)blockIdx.x * 256 + threadIdx.x; i < n;
         i += (long long)gridDim.x * 256)
        m = fmaxf(m, fabsf(x[i]));
    red[threadIdx.x] = m;
    __syncthreads();
    for (int s = 128; s > 0; s >>= 1) {
        if ((int)threadIdx.x < s)
            red[threadIdx.x] = fmaxf(red[threadIdx.x], red[threadIdx.x + s]);
        __syncthreads();
    }
    if (threadIdx.x == 0)
        atomicMax((unsigned*)slot, __float_as_uint(red[0]));
}

// x NCHW f32 -> pixel-major quantized f16: xqT[b*1024+hw][c]
__global__ __launch_bounds__(256)
void transq_x_k(const float* x, _Float16* xqT, const float* slot) {
    __shared__ float tile[64][65];
    int bid = blockIdx.x;
    int hwT = bid & 15, cT = (bid >> 4) & 15, b = bid >> 8;
    const float s  = qscale(*slot);
    const float is = 1.0f / s;
    const float* src = x + ((size_t)b * 1024 + cT * 64) * 1024 + hwT * 64;
#pragma unroll
    for (int e = 0; e < 16; ++e) {
        int idx = e * 256 + threadIdx.x;
        int cl = idx >> 6, hl = idx & 63;
        tile[cl][hl] = src[(size_t)cl * 1024 + hl];
    }
    __syncthreads();
    _Float16* dst = xqT + ((size_t)b * 1024 + hwT * 64) * 1024 + cT * 64;
#pragma unroll
    for (int e = 0; e < 16; ++e) {
        int idx = e * 256 + threadIdx.x;
        int hl = idx >> 6, cl = idx & 63;
        float v = tile[cl][hl];
        dst[(size_t)hl * 1024 + cl] = (_Float16)(rintf(v * is) * s);
    }
}

// t1 f32 [32768][256] -> padded quantized f16 [32][34][34][256]
__global__ __launch_bounds__(256)
void quant_pad_k(const float* t1, _Float16* out, const float* slot) {
    long long idx = (long long)blockIdx.x * 256 + threadIdx.x;
    const long long total = 32LL * 34 * 34 * 256;
    if (idx >= total) return;
    int c = (int)(idx & 255);
    long long r = idx >> 8;
    int pw = (int)(r % 34); r /= 34;
    int ph = (int)(r % 34);
    int b  = (int)(r / 34);
    _Float16 v = (_Float16)0.0f;
    if (ph >= 1 && ph <= 32 && pw >= 1 && pw <= 32) {
        float s = qscale(*slot);
        float x = t1[(((size_t)b * 1024) + (size_t)(ph - 1) * 32 + (pw - 1)) * 256 + c];
        v = (_Float16)(rintf(x / s) * s);
    }
    out[idx] = v;
}

__global__ __launch_bounds__(256)
void quant_act_k(const float* in, _Float16* out, const float* slot, long long n) {
    long long i = (long long)blockIdx.x * 256 + threadIdx.x;
    if (i >= n) return;
    float s = qscale(*slot);
    out[i] = (_Float16)(rintf(in[i] / s) * s);
}

// Depthwise 3x3 PEG (+bias) + BN1 + ReLU + absmax, pixel-major layout.
__global__ __launch_bounds__(256)
void peg_k(const float* h1, const float* wq, const float* pb, const float* bn1p,
           float* t1, const float* slot_in, float* slot_out) {
    int p = blockIdx.x;          // pixel
    int c = threadIdx.x;         // channel 0..255
    int b = p >> 10, h = (p >> 5) & 31, w = p & 31;
    const float s  = qscale(*slot_in);
    const float is = 1.0f / s;
    float acc = 0.0f;
#pragma unroll
    for (int dy = 0; dy < 3; ++dy)
#pragma unroll
        for (int dx = 0; dx < 3; ++dx) {
            int hh = h + dy - 1, ww = w + dx - 1;
            if (hh < 0 || hh > 31 || ww < 0 || ww > 31) continue;
            float xv = h1[((size_t)(b << 10) + (hh << 5) + ww) * 256 + c];
            xv = rintf(xv * is) * s;                 // 16-bit act quant
            acc += xv * wq[c * 9 + dy * 3 + dx];     // pow2 weight
        }
    acc = acc + pb[c];
    acc = acc * bn1p[2 * c] + bn1p[2 * c + 1];
    acc = fmaxf(acc, 0.0f);
    t1[(size_t)p * 256 + c] = acc;
    __shared__ float red[256];
    red[c] = acc;
    __syncthreads();
    for (int st = 128; st > 0; st >>= 1) {
        if (c < st) red[c] = fmaxf(red[c], red[c + st]);
        __syncthreads();
    }
    if (c == 0) atomicMax((unsigned*)slot_out, __float_as_uint(red[0]));
}

// ---------------------------------------------------------------------------
// WMMA GEMM:  D[pixel][oc] = A[pixel][K] * Bpacked[K][oc]
//   AMODE 0: A dense pixel-major [32768][K]
//   AMODE 1: A implicit im2col from padded f16 [32][34][34][256], K = 2304
//   EPI 0: store f32 [p][N] + absmax
//   EPI 1: BN + ReLU, store f32 [p][N] + absmax
//   EPI 2: BN + residual(x NCHW) + ReLU -> d_out NCHW f32
// Tiles: 128(M) x 128(N), BK=64, 8 wave32s; wave tile = 64x32 -> 8 WMMAs/step.
// ---------------------------------------------------------------------------
#define BM 128
#define BK 64
#define RS 72          /* padded LDS row stride (halves) to spread banks */
#define ABUF (BM * RS)

template<int AMODE>
__device__ __forceinline__ void load_tile_A(const _Float16* __restrict__ A,
                                            int m0, int kt, int K,
                                            _Float16* buf, int tid) {
#pragma unroll
    for (int i = 0; i < 4; ++i) {
        int c   = i * 256 + tid;        // 0..1023 chunks of 16B
        int row = c >> 3;               // 0..127
        int off = (c & 7) * 8;          // halves within row
        const _Float16* src;
        if (AMODE == 0) {
            int p = m0 + row;
            src = A + (size_t)p * K + kt * BK + off;
        } else {
            int p = m0 + row;
            int b = p >> 10, h = (p >> 5) & 31, w = p & 31;
            int k0 = kt * BK;
            int j  = k0 >> 8;           // tap 0..8 (BK divides 256, no crossing)
            int c0 = k0 & 255;
            int dy = j / 3, dx = j % 3; // padded offsets
            src = A + (((size_t)(b * 34 + h + dy) * 34) + (w + dx)) * 256 + c0 + off;
        }
        unsigned l = (unsigned)(uintptr_t)&buf[row * RS + off];
        async_cp16(l, src);
    }
}

template<int AMODE, int EPI>
__global__ __launch_bounds__(256)
void gemm_wmma_k(const _Float16* __restrict__ A,
                 const _Float16* __restrict__ Bp,
                 int K, int N,
                 float* __restrict__ Out,
                 const float* __restrict__ bnp,
                 const float* __restrict__ resid,
                 float* __restrict__ absmax_slot) {
    __shared__ __align__(16) _Float16 smem[2 * ABUF];

    const int tid  = threadIdx.x;
    const int lane = tid & 31;
    const int wave = tid >> 5;
    const int wm   = wave >> 2;     // 0..1  (64 rows each)
    const int wn   = wave & 3;      // 0..3  (32 cols each)
    const int m0   = blockIdx.x * BM;
    const int n0   = blockIdx.y * 128;
    const int KT   = K >> 6;

    v8f acc[4][2];
#pragma unroll
    for (int fi = 0; fi < 4; ++fi)
#pragma unroll
        for (int ni = 0; ni < 2; ++ni)
            acc[fi][ni] = (v8f)(0.0f);

    const int r  = lane & 15;
    const int hi = lane >> 4;

    load_tile_A<AMODE>(A, m0, 0, K, smem, tid);

    for (int kt = 0; kt < KT; ++kt) {
        wait_async0();               // my chunks of tile kt arrived
        __syncthreads();             // everyone's arrived; prev tile fully read
        _Float16* cur = smem + (kt & 1) * ABUF;
        if (kt + 1 < KT)
            load_tile_A<AMODE>(A, m0, kt + 1, K, smem + ((kt + 1) & 1) * ABUF, tid);

#pragma unroll
        for (int kk = 0; kk < 2; ++kk) {          // two 32-K steps per tile
            const int kbg = kt * 2 + kk;          // global 32-K block

            // ---- hoist all fragment loads ahead of the WMMA burst ----
            v16h bfrag[2];
#pragma unroll
            for (int ni = 0; ni < 2; ++ni) {
                int nf = n0 + wn * 32 + ni * 16;
                const _Float16* bp =
                    Bp + ((size_t)(kbg * (N >> 4) + (nf >> 4)) * 32 + lane) * 16;
                bfrag[ni] = *(const v16h*)bp;      // L2-resident packed weights
            }
            v16h afrag[4];
#pragma unroll
            for (int fi = 0; fi < 4; ++fi) {
                const _Float16* ap =
                    cur + (wm * 64 + fi * 16 + r) * RS + kk * 32 + hi * 16;
                v8h lo = *(const v8h*)ap;
                v8h hh = *(const v8h*)(ap + 8);
                afrag[fi] = cat16(lo, hh);
            }
            // ---- 8 back-to-back WMMAs on independent accumulators ----
#pragma unroll
            for (int fi = 0; fi < 4; ++fi)
#pragma unroll
                for (int ni = 0; ni < 2; ++ni)
                    acc[fi][ni] = __builtin_amdgcn_wmma_f32_16x16x32_f16(
                        false, afrag[fi], false, bfrag[ni], (short)0,
                        acc[fi][ni], false, false);
        }
    }

    // ------------------------- epilogue -------------------------
    const int ln   = lane & 15;
    const int colb = n0 + wn * 32;
    if (EPI == 0 || EPI == 1) {
        float lmax = 0.0f;
#pragma unroll
        for (int fi = 0; fi < 4; ++fi)
#pragma unroll
            for (int ni = 0; ni < 2; ++ni) {
                int oc = colb + ni * 16 + ln;
                float inv = 1.0f, beta = 0.0f;
                if (EPI == 1) { inv = bnp[2 * oc]; beta = bnp[2 * oc + 1]; }
#pragma unroll
                for (int v = 0; v < 8; ++v) {
                    int p = m0 + wm * 64 + fi * 16 + hi * 8 + v;
                    float y = acc[fi][ni][v];
                    if (EPI == 1) y = fmaxf(y * inv + beta, 0.0f);
                    Out[(size_t)p * N + oc] = y;
                    lmax = fmaxf(lmax, fabsf(y));
                }
            }
        atomicMax((unsigned*)absmax_slot, __float_as_uint(lmax));
    } else {
#pragma unroll
        for (int fi = 0; fi < 4; ++fi)
#pragma unroll
            for (int ni = 0; ni < 2; ++ni) {
                int oc = colb + ni * 16 + ln;
                float inv = bnp[2 * oc], beta = bnp[2 * oc + 1];
#pragma unroll
                for (int v = 0; v < 8; ++v) {
                    int p  = m0 + wm * 64 + fi * 16 + hi * 8 + v;
                    int b  = p >> 10;
                    int hw = p & 1023;
                    size_t o = (size_t)b * 1048576 + (size_t)oc * 1024 + hw;
                    float y = acc[fi][ni][v] * inv + beta + resid[o];
                    Out[o] = fmaxf(y, 0.0f);
                }
            }
    }
}

// ---------------------------------------------------------------------------
// Host-side launch
// ---------------------------------------------------------------------------
extern "C" void kernel_launch(void* const* d_in, const int* in_sizes, int n_in,
                              void* d_out, int out_size, void* d_ws, size_t ws_size,
                              hipStream_t stream) {
    (void)in_sizes; (void)n_in; (void)out_size; (void)ws_size;

    const float* x     = (const float*)d_in[0];
    const float* w1    = (const float*)d_in[1];
    const float* pegw  = (const float*)d_in[2];
    const float* pegb  = (const float*)d_in[3];
    const float* w2    = (const float*)d_in[4];
    const float* w3    = (const float*)d_in[5];
    const float* bn1g = (const float*)d_in[6],  *bn1b = (const float*)d_in[7];
    const float* bn1m = (const float*)d_in[8],  *bn1v = (const float*)d_in[9];
    const float* bn2g = (const float*)d_in[10], *bn2b = (const float*)d_in[11];
    const float* bn2m = (const float*)d_in[12], *bn2v = (const float*)d_in[13];
    const float* bn3g = (const float*)d_in[14], *bn3b = (const float*)d_in[15];
    const float* bn3m = (const float*)d_in[16], *bn3v = (const float*)d_in[17];
    float* out = (float*)d_out;

    // bump allocator over d_ws (256B aligned)
    char* base = (char*)d_ws;
    size_t off = 0;
    auto alloc = [&](size_t bytes) -> char* {
        char* p = base + off;
        off += (bytes + 255) & ~(size_t)255;
        return p;
    };
    float*    scales = (float*)   alloc(8 * 4);
    float*    bn1p   = (float*)   alloc(256 * 2 * 4);
    float*    bn2p   = (float*)   alloc(256 * 2 * 4);
    float*    bn3p   = (float*)   alloc(1024 * 2 * 4);
    float*    pegwq  = (float*)   alloc(256 * 9 * 4);
    _Float16* w1p    = (_Float16*)alloc((size_t)1024 * 256 * 2);
    _Float16* w2p    = (_Float16*)alloc((size_t)2304 * 256 * 2);
    _Float16* w3p    = (_Float16*)alloc((size_t)256 * 1024 * 2);
    _Float16* xqT    = (_Float16*)alloc((size_t)32768 * 1024 * 2);
    float*    h1     = (float*)   alloc((size_t)32768 * 256 * 4);  // conv1 out, reused as t2
    float*    t1     = (float*)   alloc((size_t)32768 * 256 * 4);  // PEG+BN1+ReLU out
    _Float16* t1qp   = (_Float16*)alloc((size_t)32 * 34 * 34 * 256 * 2);
    _Float16* t2q    = (_Float16*)alloc((size_t)32768 * 256 * 2);

    // -- prep --
    init_scales_k<<<1, 32, 0, stream>>>(scales);
    bn_prep_k<<<1, 256, 0, stream>>>(bn1g, bn1b, bn1m, bn1v, bn1p, 256);
    bn_prep_k<<<1, 256, 0, stream>>>(bn2g, bn2b, bn2m, bn2v, bn2p, 256);
    bn_prep_k<<<4, 256, 0, stream>>>(bn3g, bn3b, bn3m, bn3v, bn3p, 1024);
    quant_pegw_k<<<9, 256, 0, stream>>>(pegw, pegwq, 256 * 9);
    packw_k<<<1024, 256, 0, stream>>>(w1, w1p, 256, 1024, 0);
    packw_k<<<2304, 256, 0, stream>>>(w2, w2p, 256, 2304, 1);
    packw_k<<<1024, 256, 0, stream>>>(w3, w3p, 1024, 256, 0);

    // -- stage 0: activation quant scale for x, transpose to pixel-major f16 --
    absmax_k<<<2048, 256, 0, stream>>>(x, 33554432LL, &scales[0]);
    transq_x_k<<<8192, 256, 0, stream>>>(x, xqT, &scales[0]);

    // -- conv1: 1x1 1024->256 (K=1024) --
    gemm_wmma_k<0, 0><<<dim3(256, 2), 256, 0, stream>>>(
        xqT, w1p, 1024, 256, h1, nullptr, nullptr, &scales[1]);

    // -- PEG depthwise 3x3 (+bias) + BN1 + ReLU (+absmax) --
    peg_k<<<32768, 256, 0, stream>>>(h1, pegwq, pegb, bn1p, t1,
                                     &scales[1], &scales[2]);
    quant_pad_k<<<36992, 256, 0, stream>>>(t1, t1qp, &scales[2]);

    // -- conv2: 3x3 256->256, implicit im2col (K=2304), BN2+ReLU fused --
    gemm_wmma_k<1, 1><<<dim3(256, 2), 256, 0, stream>>>(
        t1qp, w2p, 2304, 256, h1 /*t2*/, bn2p, nullptr, &scales[3]);
    quant_act_k<<<32768, 256, 0, stream>>>(h1, t2q, &scales[3],
                                           (long long)32768 * 256);

    // -- conv3: 1x1 256->1024 (K=256), BN3 + residual + ReLU -> NCHW out --
    gemm_wmma_k<0, 2><<<dim3(256, 8), 256, 0, stream>>>(
        t2q, w3p, 256, 1024, out, bn3p, x, nullptr);
}
